// Decoder_60687887892701
// MI455X (gfx1250) — compile-verified, matrix operand order checked by reading, not run on hardware
//
#include <hip/hip_runtime.h>
#include <hip/hip_bf16.h>

typedef __attribute__((ext_vector_type(16))) __bf16 v16bf;
typedef __attribute__((ext_vector_type(8)))  float  v8f;
typedef __attribute__((ext_vector_type(4)))  __bf16 bf4;

#define Bsz   256
#define Tlen  256
#define Denc  1024
#define Hdim  1024

// ---------------------------------------------------------------------------
// Pack a row-major f32 weight W[K,N] into bf16 WMMA B-fragment order:
//   packed[(((kt*(N/16))+nt)*32 + lane)*16 + j]
// lane: n = nt*16 + (lane&15); j: kk = (j/8)*16 + (lane>=16?8:0) + (j%8)
// ---------------------------------------------------------------------------
__global__ __launch_bounds__(256) void pack_rowmajor(const float* __restrict__ W,
                                                     int K, int N,
                                                     __bf16* __restrict__ out,
                                                     long long total) {
    long long tid = (long long)blockIdx.x * 256 + threadIdx.x;
    if (tid >= total) return;
    int lane = (int)(tid & 31);
    long long g = tid >> 5;                 // kt*(N/16)+nt
    int ntiles = N >> 4;
    int kt = (int)(g / ntiles);
    int nt = (int)(g % ntiles);
    int n  = nt * 16 + (lane & 15);
    union { __bf16 h[16]; uint4 q[2]; } t;
#pragma unroll
    for (int j = 0; j < 16; ++j) {
        int kk = ((j >> 3) << 4) + ((lane >> 4) << 3) + (j & 7);
        int k  = (kt << 5) + kk;
        t.h[j] = (__bf16)W[(long long)k * N + n];
    }
    uint4* o = (uint4*)(out + tid * 16);
    o[0] = t.q[0];
    o[1] = t.q[1];
}

// Same packing for the concatenated transposed RNN weights:
//   Wcat[k,n] = (k<1024) ? W_ih[n,k] : W_hh[n,k-1024]      (K=2048, N=1024)
__global__ __launch_bounds__(256) void pack_cat(const float* __restrict__ Wih,
                                                const float* __restrict__ Whh,
                                                __bf16* __restrict__ out) {
    long long tid = (long long)blockIdx.x * 256 + threadIdx.x;   // exact grid
    int lane = (int)(tid & 31);
    long long g = tid >> 5;
    int nt = (int)(g & 63);                 // N/16 = 64
    int kt = (int)(g >> 6);
    int n  = nt * 16 + (lane & 15);
    union { __bf16 h[16]; uint4 q[2]; } t;
#pragma unroll
    for (int j = 0; j < 16; ++j) {
        int kk = ((j >> 3) << 4) + ((lane >> 4) << 3) + (j & 7);
        int k  = (kt << 5) + kk;
        float v = (k < Hdim) ? Wih[(long long)n * Denc + k]
                             : Whh[(long long)n * Hdim + (k - Hdim)];
        t.h[j] = (__bf16)v;
    }
    uint4* o = (uint4*)(out + tid * 16);
    o[0] = t.q[0];
    o[1] = t.q[1];
}

__global__ __launch_bounds__(256) void cvt_f32_bf16(const float* __restrict__ in,
                                                    __bf16* __restrict__ out,
                                                    long long n) {
    long long i = ((long long)blockIdx.x * 256 + threadIdx.x) * 4;
    if (i >= n) return;
    float4 v = *(const float4*)(in + i);
    union { __bf16 h[4]; uint2 q; } t;
    t.h[0] = (__bf16)v.x; t.h[1] = (__bf16)v.y;
    t.h[2] = (__bf16)v.z; t.h[3] = (__bf16)v.w;
    *(uint2*)(out + i) = t.q;
}

__global__ __launch_bounds__(256) void zero_bf16(__bf16* __restrict__ p, long long n) {
    long long i = ((long long)blockIdx.x * 256 + threadIdx.x) * 8;
    if (i >= n) return;
    uint4 z; z.x = 0; z.y = 0; z.z = 0; z.w = 0;
    *(uint4*)(p + i) = z;
}

// ---------------------------------------------------------------------------
// WMMA bf16 GEMM:  out = act( [A0|A1] @ Wpacked + bias0 (+bias1) )
//   workgroup tile 64(M) x 128(N): 8 waves, each wave 16x64 (4 wmma tiles)
//   Packed-W K-tiles (8 KB) staged to LDS with global_load_async_to_lds_b128,
//   double-buffered on ASYNCcnt; WMMA B operands fed from LDS (ds_load_b128).
// ---------------------------------------------------------------------------
__global__ __launch_bounds__(256) void gemm_wmma_bf16(
    const __bf16* __restrict__ A0, long long ldA0,
    const __bf16* __restrict__ A1, long long ldA1, int ksplit,
    const __bf16* __restrict__ Wp,
    const float* __restrict__ bias0, const float* __restrict__ bias1,
    int K, int N,
    float* __restrict__ outF, long long ldF,
    __bf16* __restrict__ outB, long long ldB,
    int act_tanh) {
    // 2 buffers x (8 ntiles x 32 lanes x 16 bf16) = 2 x 8 KB
    __shared__ __attribute__((aligned(16))) __bf16 smem[2][8 * 32 * 16];

    const int tid  = threadIdx.x;
    const int wave = tid >> 5;
    const int lane = tid & 31;
    const int wm = wave >> 1;                 // 0..3 (M subtile)
    const int wn = wave & 1;                  // 0..1 (N half)
    const int rowBase = blockIdx.x * 64 + wm * 16;
    const int colBase = blockIdx.y * 128 + wn * 64;
    const int mrow = rowBase + (lane & 15);   // A row this lane holds
    const int off  = (lane >> 4) << 3;        // 0 or 8 (K sub-offset)
    const int ntiles = N >> 4;
    const int nkt = K >> 5;

    // staging assignment: thread tid copies fragment (nt = blk0 + tid/32, lane)
    const long long fragIdx = ((long long)blockIdx.y * 8 + (tid >> 5)) * 32 + (tid & 31);
    // generic->LDS: low 32 bits of the flat address are the LDS byte address
    const unsigned ldsDst[2] = {
        (unsigned)(uintptr_t)(const void*)&smem[0][0] + (unsigned)tid * 32u,
        (unsigned)(uintptr_t)(const void*)&smem[1][0] + (unsigned)tid * 32u };

    v8f acc[4] = {};

    // issue async copy of packed-W K-tile `kt` into LDS buffer `buf`
    auto issue_tile = [&](int kt, int buf) {
        unsigned long long g =
            (unsigned long long)(uintptr_t)(Wp + ((long long)kt * ntiles * 32 + fragIdx) * 16);
        unsigned l = ldsDst[buf];
        asm volatile("global_load_async_to_lds_b128 %0, %1, off\n\t"
                     "global_load_async_to_lds_b128 %0, %1, off offset:16"
                     :: "v"(l), "v"(g) : "memory");
    };

    auto chunk = [&](int kt, const __bf16* aChunk) {
        // ---- A fragment (ISA 16-bit 16x32 layout): two 16B halves ----
        union { v16bf v; uint4 q[2]; } a;
        a.q[0] = *(const uint4*)(aChunk + off);        // K 0..7  / 8..15
        a.q[1] = *(const uint4*)(aChunk + 16 + off);   // K 16..23 / 24..31
        const int buf = kt & 1;
        if (kt + 1 < nkt) {
            issue_tile(kt + 1, buf ^ 1);
            asm volatile("s_wait_asynccnt 0x2" ::: "memory");  // current tile done
        } else {
            asm volatile("s_wait_asynccnt 0x0" ::: "memory");
        }
        __syncthreads();                                        // tile visible to all
        const __bf16* sb = &smem[buf][(wn * 4 * 32 + lane) * 16];
#pragma unroll
        for (int s = 0; s < 4; ++s) {
            union { v16bf v; uint4 q[2]; } b;
            const uint4* bp = (const uint4*)(sb + s * (32 * 16));  // ds_load_b128 x2
            b.q[0] = bp[0];
            b.q[1] = bp[1];
            acc[s] = __builtin_amdgcn_wmma_f32_16x16x32_bf16(
                false, a.v, false, b.v, (short)0, acc[s], false, false);
        }
        __syncthreads();                    // all waves done before buf is refilled
    };

    issue_tile(0, 0);
    const __bf16* aRow0 = A0 + (long long)mrow * ldA0;
    const __bf16* aRow1 = A1 + (long long)mrow * ldA1;
    int kt = 0;
    const int nkt0 = ksplit >> 5;           // ksplit is a multiple of 32
    for (; kt < nkt0; ++kt) chunk(kt, aRow0 + (kt << 5));
    for (; kt < nkt;  ++kt) chunk(kt, aRow1 + ((kt << 5) - ksplit));

    // ---- epilogue: bias + optional tanh, f32 and/or bf16 stores ----
#pragma unroll
    for (int s = 0; s < 4; ++s) {
        int col = colBase + s * 16 + (lane & 15);
        float bs = bias0 ? bias0[col] : 0.0f;
        if (bias1) bs += bias1[col];
#pragma unroll
        for (int r = 0; r < 8; ++r) {
            int row = rowBase + r + ((lane >> 4) << 3);
            float v = acc[s][r] + bs;
            if (act_tanh) v = tanhf(v);
            if (outF) outF[(long long)row * ldF + col] = v;
            if (outB) outB[(long long)row * ldB + col] = (__bf16)v;
        }
    }
}

// ---------------------------------------------------------------------------
// Row softmax over T=256 logits; one block (256 thr) per row.
// ---------------------------------------------------------------------------
__global__ __launch_bounds__(256) void softmax_rows(const float* __restrict__ logits,
                                                    float* __restrict__ attn) {
    __shared__ float red[8];
    __shared__ float bro[1];
    const int row = blockIdx.x;
    const int tid = threadIdx.x;
    float v = logits[(long long)row * Tlen + tid];

    float m = v;
#pragma unroll
    for (int o = 16; o > 0; o >>= 1) m = fmaxf(m, __shfl_xor(m, o, 32));
    if ((tid & 31) == 0) red[tid >> 5] = m;
    __syncthreads();
    if (tid == 0) {
        float t = red[0];
#pragma unroll
        for (int i = 1; i < 8; ++i) t = fmaxf(t, red[i]);
        bro[0] = t;
    }
    __syncthreads();
    m = bro[0];

    float e = __expf(v - m);
    float s = e;
#pragma unroll
    for (int o = 16; o > 0; o >>= 1) s += __shfl_xor(s, o, 32);
    __syncthreads();
    if ((tid & 31) == 0) red[tid >> 5] = s;
    __syncthreads();
    if (tid == 0) {
        float t = red[0];
#pragma unroll
        for (int i = 1; i < 8; ++i) t += red[i];
        bro[0] = t;
    }
    __syncthreads();
    attn[(long long)row * Tlen + tid] = e / bro[0];
}

// ---------------------------------------------------------------------------
// Context: c[b,d] = sum_t attn[b,t] * h_bf[b,t,d]  -> bf16.  Memory-bound:
// one block per batch row, 256 threads x 4 d each (coalesced bf16x4 loads).
// ---------------------------------------------------------------------------
__global__ __launch_bounds__(256) void context_kernel(const __bf16* __restrict__ hbf,
                                                      const float* __restrict__ attn,
                                                      __bf16* __restrict__ cbf) {
    const int b = blockIdx.x;
    const int d = threadIdx.x * 4;
    const __bf16* hb = hbf + ((long long)b * Tlen) * Denc + d;
    const float* ab = attn + (long long)b * Tlen;
    float a0 = 0.f, a1 = 0.f, a2 = 0.f, a3 = 0.f;
    for (int t = 0; t < Tlen; ++t) {
        float w = ab[t];                       // uniform -> scalar load
        bf4 hv = *(const bf4*)(hb + (long long)t * Denc);
        if (t + 8 < Tlen)
            __builtin_prefetch((const void*)(hb + (long long)(t + 8) * Denc), 0, 1);
        a0 += w * (float)hv.x;
        a1 += w * (float)hv.y;
        a2 += w * (float)hv.z;
        a3 += w * (float)hv.w;
    }
    union { __bf16 h[4]; uint2 q; } o;
    o.h[0] = (__bf16)a0; o.h[1] = (__bf16)a1;
    o.h[2] = (__bf16)a2; o.h[3] = (__bf16)a3;
    *(uint2*)(cbf + (long long)b * Denc + d) = o.q;
}

// ---------------------------------------------------------------------------
extern "C" void kernel_launch(void* const* d_in, const int* in_sizes, int n_in,
                              void* d_out, int out_size, void* d_ws, size_t ws_size,
                              hipStream_t stream) {
    const float* h   = (const float*)d_in[0];
    const float* W1  = (const float*)d_in[1];
    const float* b1  = (const float*)d_in[2];
    const float* W2  = (const float*)d_in[3];
    const float* b2  = (const float*)d_in[4];
    const float* Wih = (const float*)d_in[5];
    const float* bih = (const float*)d_in[6];
    const float* Whh = (const float*)d_in[7];
    const float* bhh = (const float*)d_in[8];
    float* out = (float*)d_out;

    char* ws = (char*)d_ws;
    auto alloc = [&](size_t bytes) {
        char* p = ws;
        ws += (bytes + 255) & ~(size_t)255;
        return p;
    };
    __bf16* W1p    = (__bf16*)alloc((size_t)2048 * 1024 * 2);
    __bf16* W2p    = (__bf16*)alloc((size_t)1024 * 256 * 2);
    __bf16* Wcp    = (__bf16*)alloc((size_t)2048 * 1024 * 2);
    __bf16* hbf    = (__bf16*)alloc((size_t)Bsz * Tlen * Denc * 2);
    __bf16* zbf    = (__bf16*)alloc((size_t)Bsz * Hdim * 2);
    float*  logit  = (float*) alloc((size_t)Bsz * Tlen * 4);
    float*  attn   = (float*) alloc((size_t)Bsz * Tlen * 4);
    __bf16* cbf    = (__bf16*)alloc((size_t)Bsz * Denc * 2);
    __bf16* sbuf0  = (__bf16*)alloc((size_t)Bsz * Hdim * 2);
    __bf16* sbuf1  = (__bf16*)alloc((size_t)Bsz * Hdim * 2);

    // ---- one-time (per call) preprocessing ----
    pack_rowmajor<<<512, 256, 0, stream>>>(W1, 2048, 1024, W1p, 131072);
    pack_rowmajor<<<64,  256, 0, stream>>>(W2, 1024, 256,  W2p, 16384);
    pack_cat<<<512, 256, 0, stream>>>(Wih, Whh, Wcp);
    cvt_f32_bf16<<<65536, 256, 0, stream>>>(h, hbf, (long long)Bsz * Tlen * Denc);
    zero_bf16<<<128, 256, 0, stream>>>(sbuf0, (long long)Bsz * Hdim);

    __bf16* sin  = sbuf0;
    __bf16* sout = sbuf1;

    for (int i = 0; i < Tlen; ++i) {
        // z = tanh([s, h_t] @ W1 + b1)   -> bf16
        gemm_wmma_bf16<<<dim3(4, 8), 256, 0, stream>>>(
            sin, (long long)Hdim,
            hbf + (long long)i * Denc, (long long)Tlen * Denc, Hdim,
            W1p, b1, nullptr, 2048, 1024,
            nullptr, 0, zbf, (long long)Hdim, 1);
        // logits = z @ W2 + b2           -> f32
        gemm_wmma_bf16<<<dim3(4, 2), 256, 0, stream>>>(
            zbf, (long long)Hdim, zbf, (long long)Hdim, 1024,
            W2p, b2, nullptr, 1024, 256,
            logit, (long long)Tlen, nullptr, 0, 0);
        // a = softmax(logits)
        softmax_rows<<<Bsz, 256, 0, stream>>>(logit, attn);
        // c[b,:] = h[b]^T a[b]           -> bf16  (L2-resident h, bf16 traffic)
        context_kernel<<<Bsz, 256, 0, stream>>>(hbf, attn, cbf);
        // s' = tanh([c, s] @ [Wih^T;Whh^T] + b_ih + b_hh); write out[:,i,:] f32
        gemm_wmma_bf16<<<dim3(4, 8), 256, 0, stream>>>(
            cbf, (long long)Denc, sin, (long long)Hdim, Denc,
            Wcp, bih, bhh, 2048, 1024,
            out + (long long)i * Hdim, (long long)Tlen * Hdim,
            sout, (long long)Hdim, 1);
        // swap state double-buffer
        __bf16* tmp = sin; sin = sout; sout = tmp;
    }
}
